// SRECG_68307159875654
// MI455X (gfx1250) — matrix operand order repaired and checked by reading, not compiled, not run on hardware
//
#include <hip/hip_runtime.h>
#include <hip/hip_bf16.h>
#include <cstdint>
#include <cstddef>

typedef __attribute__((ext_vector_type(16))) _Float16 v16h;
typedef __attribute__((ext_vector_type(8)))  _Float16 v8h;
typedef __attribute__((ext_vector_type(8)))  float    v8f;

constexpr int BATCH  = 16;
constexpr int TT     = 2048;
constexpr int TLEN5  = TT * 5;
constexpr int TLEN10 = TT * 10;

__device__ __forceinline__ float leakyf(float v) { return v >= 0.f ? v : 0.01f * v; }
__device__ __forceinline__ float siluf(float v)  { return v / (1.f + __expf(-v)); }

enum { EPI_F32 = 0, EPI_LEAKY_F32 = 1, EPI_RESID_F32_F16 = 2, EPI_SHUF_LEAKY_F16 = 3 };

// K layout helpers: each input channel occupies KSLOT entries of K.
// KW==1 -> KSLOT=1 (pure GEMM). KW>1 -> KSLOT=16 (kw 0..KW-1 valid, rest zero wt).
constexpr int kslot_of(int KW)                { return KW == 1 ? 1 : 16; }
constexpr int nch_of(int CIN, int CIC)        { return (CIN + CIC - 1) / CIC; }
constexpr int ktotp_of(int CIN, int KW, int CIC) { return nch_of(CIN, CIC) * CIC * kslot_of(KW); }
constexpr int coutp_of(int COUT)              { return (COUT + 15) & ~15; }

// ---------------------------------------------------------------------------
// Weight prep: f32 (COUT,CIN,KW) -> f16 (COUTP, KTOTP) in K-slot layout,
// zero padded so conv fragments never need guards.
// ---------------------------------------------------------------------------
template <int CIN, int COUT, int KW, int CIC>
__global__ void prep_w_k(const float* __restrict__ w, _Float16* __restrict__ w16) {
    constexpr int KTOTP = ktotp_of(CIN, KW, CIC);
    constexpr int COUTP = coutp_of(COUT);
    int i = blockIdx.x * blockDim.x + threadIdx.x;
    if (i >= COUTP * KTOTP) return;
    int co = i / KTOTP;
    int k  = i - co * KTOTP;
    float v = 0.f;
    if (co < COUT) {
        if (KW == 1) {
            if (k < CIN) v = w[(size_t)co * CIN + k];
        } else {
            int ci = k >> 4, kw = k & 15;
            if (ci < CIN && kw < KW) v = w[((size_t)co * CIN + ci) * KW + kw];
        }
    }
    w16[i] = (_Float16)v;
}

// ---------------------------------------------------------------------------
// Implicit-GEMM conv1d via V_WMMA_F32_16X16X32_F16 (wave32).
// A = weights (16 cout x 32 k) loaded straight from prepped global f16,
// B = im2col input tile staged in LDS. 4 waves/block, 64 time cols/block.
// ---------------------------------------------------------------------------
template <int CIN, int COUT, int KW, int PAD, int CIC, int EPI, int R>
__global__ void __launch_bounds__(128)
conv1d_wmma(const _Float16* __restrict__ xin,   // (b, CIN, T) f16
            const _Float16* __restrict__ w16,   // (COUTP, KTOTP) f16 prepped
            const float*    __restrict__ bias,  // (COUT) or nullptr
            float*          __restrict__ outf,
            _Float16*       __restrict__ outh,
            const float*    __restrict__ resid,
            int T)
{
    constexpr int TILE_N = 64;
    constexpr int BLK    = 128;
    constexpr int KSLOT  = kslot_of(KW);
    constexpr int KCP    = CIC * KSLOT;               // per-chunk K (mult of 32)
    constexpr int KTOTP  = ktotp_of(CIN, KW, CIC);
    constexpr int XW     = TILE_N + KW - 1;
    constexpr int XWP    = (KW == 1) ? CIC : ((XW + 7) & ~7);   // 16B-aligned rows
    constexpr int LDSN   = (KW == 1) ? (TILE_N * CIC) : (CIC * XWP);

    __shared__ _Float16 Xs[LDSN];

    const int tid      = threadIdx.x;
    const int lane     = tid & 31;
    const int wave     = tid >> 5;
    const int t0       = blockIdx.x * TILE_N;
    const int coutBase = blockIdx.y * 16;
    const int b        = blockIdx.z;

    const int mrow = lane & 15;
    const int nl   = wave * 16 + (lane & 15);
    const int hiA0 = (lane < 16) ? 0 : 16;    // A 16x32 f16 fragment layout
    const int hiA1 = (lane < 16) ? 8 : 24;
    const int hiB  = (lane < 16) ? 0 : 16;    // B 32x16 f16 fragment layout

    const _Float16* xb   = xin + (size_t)b * CIN * T;
    const _Float16* wrow = w16 + (size_t)(coutBase + mrow) * KTOTP;

    v8f acc = {};

    for (int ciBase = 0; ciBase < CIN; ciBase += CIC) {
        __syncthreads();
        if (KW == 1) {
            // stage time-major Xst[nl][ci]; vectorized global reads (8 halves)
            for (int i = tid; i < CIC * (TILE_N / 8); i += BLK) {
                int ci  = i >> 3;
                int seg = i & 7;
                int cig = ciBase + ci;
                v8h v = {};
                if (cig < CIN)
                    v = *(const v8h*)(xb + (size_t)cig * T + t0 + seg * 8);
#pragma unroll
                for (int j = 0; j < 8; j++) Xs[(seg * 8 + j) * CIC + ci] = v[j];
            }
        } else {
            // stage channel-major halo tile Xs[ci][0..XWP)
            for (int i = tid; i < CIC * XWP; i += BLK) {
                int ci  = i / XWP;
                int j   = i - ci * XWP;
                int cig = ciBase + ci;
                int t   = t0 - PAD + j;
                _Float16 v = (_Float16)0.f;
                if (cig < CIN && j < XW && t >= 0 && t < T)
                    v = xb[(size_t)cig * T + t];
                Xs[i] = v;
            }
        }
        __syncthreads();

        const int kOff = ciBase * KSLOT;    // chunk offset in W16 K axis
#pragma unroll
        for (int ks = 0; ks < KCP; ks += 32) {
            // A fragment: two aligned 16B global loads from L2-resident weights
            v8h a0 = *(const v8h*)(wrow + kOff + ks + hiA0);
            v8h a1 = *(const v8h*)(wrow + kOff + ks + hiA1);
            v16h afr = __builtin_shufflevector(a0, a1, 0, 1, 2, 3, 4, 5, 6, 7,
                                               8, 9, 10, 11, 12, 13, 14, 15);
            v16h bfr;
            if (KW == 1) {
                // contiguous, 16B-aligned: two ds_load_b128
                const _Float16* xr = &Xs[nl * CIC + ks + hiB];
                v8h b0 = *(const v8h*)(xr);
                v8h b1 = *(const v8h*)(xr + 8);
                bfr = __builtin_shufflevector(b0, b1, 0, 1, 2, 3, 4, 5, 6, 7,
                                              8, 9, 10, 11, 12, 13, 14, 15);
            } else {
                // one channel per lane-half; contiguous 16-half window (slot 15
                // pairs with a zero weight, so its value is don't-care)
                int ci = (ks + hiB) >> 4;
                const _Float16* xr = &Xs[ci * XWP + nl];
#pragma unroll
                for (int e = 0; e < 16; e++) bfr[e] = xr[e];
            }
            acc = __builtin_amdgcn_wmma_f32_16x16x32_f16(
                false, afr, false, bfr, (short)0, acc, false, false);
        }
    }

    // epilogue: C/D layout -> VGPR r holds M=r (lanes 0-15) / M=8+r (lanes 16-31)
    const int mOff = (lane < 16) ? 0 : 8;
    const int tl   = t0 + wave * 16 + (lane & 15);
#pragma unroll
    for (int r = 0; r < 8; r++) {
        int co = coutBase + mOff + r;
        if (co >= COUT) continue;
        float v = acc[r];
        if (bias) v += bias[co];
        if (EPI == EPI_F32) {
            outf[((size_t)b * COUT + co) * T + tl] = v;
        } else if (EPI == EPI_LEAKY_F32) {
            outf[((size_t)b * COUT + co) * T + tl] = leakyf(v);
        } else if (EPI == EPI_RESID_F32_F16) {
            v += resid[((size_t)b * COUT + co) * T + tl];
            outf[((size_t)b * COUT + co) * T + tl] = v;
            outh[((size_t)b * COUT + co) * T + tl] = (_Float16)v;
        } else { // EPI_SHUF_LEAKY_F16: pixel_shuffle_1d(conv_out, R) + leaky
            int c  = co / R;
            int rr = co - c * R;
            outh[((size_t)b * (COUT / R) + c) * ((size_t)T * R) + (size_t)tl * R + rr] =
                (_Float16)leakyf(v);
        }
    }
}

// ---------------------------------------------------------------------------
// Elementwise / small kernels
// ---------------------------------------------------------------------------
__global__ void cast_f32_f16_k(const float* __restrict__ in, _Float16* __restrict__ out, int n) {
    int i = blockIdx.x * blockDim.x + threadIdx.x;
    if (i < n) out[i] = (_Float16)in[i];
}

__global__ void rmsnorm_k(const float* __restrict__ h, const float* __restrict__ w,
                          _Float16* __restrict__ xn, int T) {
    int idx = blockIdx.x * blockDim.x + threadIdx.x;
    if (idx >= BATCH * T) return;
    int b = idx / T, t = idx - b * T;
    const float* p = h + (size_t)b * 64 * T + t;
    float ss = 0.f;
    for (int d = 0; d < 64; d++) { float v = p[(size_t)d * T]; ss += v * v; }
    float inv = rsqrtf(ss * (1.f / 64.f) + 1e-5f);
    for (int d = 0; d < 64; d++)
        xn[((size_t)b * 64 + d) * T + t] = (_Float16)(p[(size_t)d * T] * inv * w[d]);
}

__global__ void dwconv_silu_k(const float* __restrict__ xz, const float* __restrict__ w,
                              const float* __restrict__ bias, float* __restrict__ xms,
                              _Float16* __restrict__ xmsh, int T) {
    long i = (long)blockIdx.x * blockDim.x + threadIdx.x;
    if (i >= (long)BATCH * 128 * T) return;
    int  t  = (int)(i % T);
    long bd = i / T;
    int  d  = (int)(bd & 127);
    int  b  = (int)(bd >> 7);
    const float* xm = xz + ((size_t)b * 256 + d) * T;
    float a = bias[d];
#pragma unroll
    for (int k = 0; k < 4; k++) {
        int tt = t + k - 3;
        if (tt >= 0) a += w[d * 4 + k] * xm[tt];
    }
    float v = siluf(a);
    size_t o = ((size_t)b * 128 + d) * T + t;
    xms[o]  = v;
    xmsh[o] = (_Float16)v;
}

__global__ void dtproj_k(const float* __restrict__ xdbl, const float* __restrict__ dtw,
                         const float* __restrict__ dtb, float* __restrict__ delta, int T) {
    long i = (long)blockIdx.x * blockDim.x + threadIdx.x;
    if (i >= (long)BATCH * 128 * T) return;
    int  t  = (int)(i % T);
    long bd = i / T;
    int  d  = (int)(bd & 127);
    int  b  = (int)(bd >> 7);
    const float* xd = xdbl + (size_t)b * 36 * T + t;
    float a = dtb[d];
#pragma unroll
    for (int r = 0; r < 4; r++) a += dtw[d * 4 + r] * xd[(size_t)r * T];
    float v = (a > 20.f) ? a : log1pf(__expf(a));
    delta[((size_t)b * 128 + d) * T + t] = v;
}

__global__ void scan_k(const float* __restrict__ u, const float* __restrict__ delta,
                       const float* __restrict__ xdbl, const float* __restrict__ A_log,
                       const float* __restrict__ Dp, float* __restrict__ y, int T) {
    int gid = blockIdx.x * blockDim.x + threadIdx.x;
    int grp = gid >> 4;         // b*128 + d
    int n   = gid & 15;
    int b   = grp >> 7;
    int d   = grp & 127;
    if (b >= BATCH) return;
    float A  = -__expf(A_log[d * 16 + n]);
    const float* Brow = xdbl + ((size_t)b * 36 + 4 + n) * T;
    const float* Crow = xdbl + ((size_t)b * 36 + 20 + n) * T;
    const float* urow = u + ((size_t)b * 128 + d) * T;
    const float* drow = delta + ((size_t)b * 128 + d) * T;
    float* yrow = y + ((size_t)b * 128 + d) * T;
    float Dd = Dp[d];
    float h  = 0.f;
    for (int t = 0; t < T; t++) {
        float dt = drow[t], ut = urow[t];
        h = h * __expf(dt * A) + (dt * ut) * Brow[t];
        float p = h * Crow[t];
        p += __shfl_xor(p, 1, 32);
        p += __shfl_xor(p, 2, 32);
        p += __shfl_xor(p, 4, 32);
        p += __shfl_xor(p, 8, 32);
        if (n == 0) yrow[t] = p + ut * Dd;
    }
}

__global__ void gate_k(const float* __restrict__ y, const float* __restrict__ xz,
                       _Float16* __restrict__ yh, int T) {
    long i = (long)blockIdx.x * blockDim.x + threadIdx.x;
    if (i >= (long)BATCH * 128 * T) return;
    int  t  = (int)(i % T);
    long bd = i / T;
    int  d  = (int)(bd & 127);
    int  b  = (int)(bd >> 7);
    float zv = xz[((size_t)b * 256 + 128 + d) * T + t];
    size_t o = ((size_t)b * 128 + d) * T + t;
    yh[o] = (_Float16)(y[o] * siluf(zv));
}

__global__ void bn_stats_k(const float* __restrict__ mtmp, float* __restrict__ stats, int T) {
    __shared__ float ssum[256];
    __shared__ float ssq[256];
    int c = blockIdx.x;
    float s = 0.f, q = 0.f;
    for (int i = threadIdx.x; i < BATCH * T; i += 256) {
        int b = i / T, t = i - b * T;
        float v = mtmp[((size_t)b * 64 + c) * T + t];
        s += v; q += v * v;
    }
    ssum[threadIdx.x] = s; ssq[threadIdx.x] = q;
    __syncthreads();
    for (int st = 128; st > 0; st >>= 1) {
        if (threadIdx.x < st) {
            ssum[threadIdx.x] += ssum[threadIdx.x + st];
            ssq[threadIdx.x]  += ssq[threadIdx.x + st];
        }
        __syncthreads();
    }
    if (threadIdx.x == 0) {
        float inv = 1.f / (float)(BATCH * T);
        float mu  = ssum[0] * inv;
        stats[c]      = mu;
        stats[64 + c] = ssq[0] * inv - mu * mu;
    }
}

__global__ void bn_apply_k(const float* __restrict__ mtmp, const float* __restrict__ stats,
                           const float* __restrict__ g, const float* __restrict__ bb,
                           const float* __restrict__ hold, _Float16* __restrict__ mh, int T) {
    long i = (long)blockIdx.x * blockDim.x + threadIdx.x;
    if (i >= (long)BATCH * 64 * T) return;
    int c = (int)((i / T) & 63);
    float v = (mtmp[i] - stats[c]) * rsqrtf(stats[64 + c] + 1e-5f) * g[c] + bb[c] + hold[i];
    mh[i] = (_Float16)v;
}

// ---------------------------------------------------------------------------
// Host launch
// ---------------------------------------------------------------------------
extern "C" void kernel_launch(void* const* d_in, const int* in_sizes, int n_in,
                              void* d_out, int out_size, void* d_ws, size_t ws_size,
                              hipStream_t stream) {
    const float* x         = (const float*)d_in[0];
    const float* conv_in_w = (const float*)d_in[1];
    const float* conv_in_b = (const float*)d_in[2];
    const float* norm_w    = (const float*)d_in[3];
    const float* in_proj_w = (const float*)d_in[4];
    const float* dwconv_w  = (const float*)d_in[5];
    const float* dwconv_b  = (const float*)d_in[6];
    const float* x_proj_w  = (const float*)d_in[7];
    const float* dt_proj_w = (const float*)d_in[8];
    const float* dt_proj_b = (const float*)d_in[9];
    const float* A_log     = (const float*)d_in[10];
    const float* Dvec      = (const float*)d_in[11];
    const float* out_proj_w= (const float*)d_in[12];
    const float* merge_w   = (const float*)d_in[13];
    const float* merge_b   = (const float*)d_in[14];
    const float* bn_g      = (const float*)d_in[15];
    const float* bn_b      = (const float*)d_in[16];
    const float* up1_w     = (const float*)d_in[17];
    const float* up1_b     = (const float*)d_in[18];
    const float* up2_w     = (const float*)d_in[19];
    const float* up2_b     = (const float*)d_in[20];
    const float* out_w     = (const float*)d_in[21];
    const float* out_b     = (const float*)d_in[22];

    char* ws = (char*)d_ws;
    size_t off = 0;
    auto alloc = [&](size_t bytes) {
        char* p = ws + off;
        off += (bytes + 255) & ~(size_t)255;
        return (void*)p;
    };

    _Float16* x_h   = (_Float16*)alloc((size_t)BATCH * 12 * TT * 2);
    float*    h     = (float*)   alloc((size_t)BATCH * 64 * TT * 4);
    _Float16* xn_h  = (_Float16*)alloc((size_t)BATCH * 64 * TT * 2);
    float*    xz    = (float*)   alloc((size_t)BATCH * 256 * TT * 4);
    float*    xms   = (float*)   alloc((size_t)BATCH * 128 * TT * 4);
    _Float16* xms_h = (_Float16*)alloc((size_t)BATCH * 128 * TT * 2);
    float*    xdbl  = (float*)   alloc((size_t)BATCH * 36 * TT * 4);
    float*    delta = (float*)   alloc((size_t)BATCH * 128 * TT * 4);
    float*    yb    = (float*)   alloc((size_t)BATCH * 128 * TT * 4);
    _Float16* yh    = (_Float16*)alloc((size_t)BATCH * 128 * TT * 2);
    float*    h2    = (float*)   alloc((size_t)BATCH * 64 * TT * 4);
    _Float16* h2_h  = (_Float16*)alloc((size_t)BATCH * 64 * TT * 2);
    float*    mtmp  = (float*)   alloc((size_t)BATCH * 64 * TT * 4);
    float*    stats = (float*)   alloc(128 * 4);
    _Float16* m_h   = (_Float16*)alloc((size_t)BATCH * 64 * TT * 2);
    _Float16* u5h   = (_Float16*)alloc((size_t)BATCH * 320 * TLEN5 * 2);
    _Float16* u10h  = (_Float16*)alloc((size_t)BATCH * 128 * TLEN10 * 2);

    // prepped f16 weights (padded cout rows x K-slot layout)
    _Float16* w16_cin   = (_Float16*)alloc((size_t)coutp_of(64)   * ktotp_of(12,  15, 16) * 2);
    _Float16* w16_inp   = (_Float16*)alloc((size_t)coutp_of(256)  * ktotp_of(64,  1,  32) * 2);
    _Float16* w16_xp    = (_Float16*)alloc((size_t)coutp_of(36)   * ktotp_of(128, 1,  32) * 2);
    _Float16* w16_op    = (_Float16*)alloc((size_t)coutp_of(64)   * ktotp_of(128, 1,  32) * 2);
    _Float16* w16_mrg   = (_Float16*)alloc((size_t)coutp_of(64)   * ktotp_of(64,  15, 16) * 2);
    _Float16* w16_up1   = (_Float16*)alloc((size_t)coutp_of(1600) * ktotp_of(64,  15, 16) * 2);
    _Float16* w16_up2   = (_Float16*)alloc((size_t)coutp_of(256)  * ktotp_of(320, 15, 16) * 2);
    _Float16* w16_out   = (_Float16*)alloc((size_t)coutp_of(12)   * ktotp_of(128, 15, 16) * 2);
    (void)ws_size; (void)in_sizes; (void)n_in; (void)out_size;

    const int EW = 256;
    auto blocks = [&](long n) { return (int)((n + EW - 1) / EW); };
    auto pgrid  = [&](long n) { return (int)((n + EW - 1) / EW); };

    // weight prep (cheap, once per launch)
    prep_w_k<12, 64, 15, 16><<<pgrid((long)coutp_of(64) * ktotp_of(12, 15, 16)), EW, 0, stream>>>(conv_in_w, w16_cin);
    prep_w_k<64, 256, 1, 32><<<pgrid((long)coutp_of(256) * ktotp_of(64, 1, 32)), EW, 0, stream>>>(in_proj_w, w16_inp);
    prep_w_k<128, 36, 1, 32><<<pgrid((long)coutp_of(36) * ktotp_of(128, 1, 32)), EW, 0, stream>>>(x_proj_w, w16_xp);
    prep_w_k<128, 64, 1, 32><<<pgrid((long)coutp_of(64) * ktotp_of(128, 1, 32)), EW, 0, stream>>>(out_proj_w, w16_op);
    prep_w_k<64, 64, 15, 16><<<pgrid((long)coutp_of(64) * ktotp_of(64, 15, 16)), EW, 0, stream>>>(merge_w, w16_mrg);
    prep_w_k<64, 1600, 15, 16><<<pgrid((long)coutp_of(1600) * ktotp_of(64, 15, 16)), EW, 0, stream>>>(up1_w, w16_up1);
    prep_w_k<320, 256, 15, 16><<<pgrid((long)coutp_of(256) * ktotp_of(320, 15, 16)), EW, 0, stream>>>(up2_w, w16_up2);
    prep_w_k<128, 12, 15, 16><<<pgrid((long)coutp_of(12) * ktotp_of(128, 15, 16)), EW, 0, stream>>>(out_w, w16_out);

    // 0) cast input to f16
    cast_f32_f16_k<<<blocks((long)BATCH * 12 * TT), EW, 0, stream>>>(x, x_h, BATCH * 12 * TT);

    // 1) conv_in 12->64 k15 + leaky
    conv1d_wmma<12, 64, 15, 7, 16, EPI_LEAKY_F32, 1>
        <<<dim3(TT / 64, 4, BATCH), 128, 0, stream>>>(x_h, w16_cin, conv_in_b, h, nullptr, nullptr, TT);

    // 2) rmsnorm -> f16
    rmsnorm_k<<<blocks((long)BATCH * TT), EW, 0, stream>>>(h, norm_w, xn_h, TT);

    // 3) in_proj 64->256 (GEMM)
    conv1d_wmma<64, 256, 1, 0, 32, EPI_F32, 1>
        <<<dim3(TT / 64, 16, BATCH), 128, 0, stream>>>(xn_h, w16_inp, nullptr, xz, nullptr, nullptr, TT);

    // 4) causal depthwise conv + SiLU
    dwconv_silu_k<<<blocks((long)BATCH * 128 * TT), EW, 0, stream>>>(xz, dwconv_w, dwconv_b, xms, xms_h, TT);

    // 5) x_proj 128->36 (GEMM)
    conv1d_wmma<128, 36, 1, 0, 32, EPI_F32, 1>
        <<<dim3(TT / 64, 3, BATCH), 128, 0, stream>>>(xms_h, w16_xp, nullptr, xdbl, nullptr, nullptr, TT);

    // 6) dt_proj + softplus
    dtproj_k<<<blocks((long)BATCH * 128 * TT), EW, 0, stream>>>(xdbl, dt_proj_w, dt_proj_b, delta, TT);

    // 7) selective scan
    scan_k<<<(BATCH * 128 * 16) / 256, 256, 0, stream>>>(xms, delta, xdbl, A_log, Dvec, yb, TT);

    // 8) gate y * silu(z)
    gate_k<<<blocks((long)BATCH * 128 * TT), EW, 0, stream>>>(yb, xz, yh, TT);

    // 9) out_proj 128->64 + residual (h) -> h2 (f32 + f16)
    conv1d_wmma<128, 64, 1, 0, 32, EPI_RESID_F32_F16, 1>
        <<<dim3(TT / 64, 4, BATCH), 128, 0, stream>>>(yh, w16_op, nullptr, h2, h2_h, h, TT);

    // 10) merge conv 64->64 k15
    conv1d_wmma<64, 64, 15, 7, 16, EPI_F32, 1>
        <<<dim3(TT / 64, 4, BATCH), 128, 0, stream>>>(h2_h, w16_mrg, merge_b, mtmp, nullptr, nullptr, TT);

    // 11) batchnorm stats + apply (+hold) -> m (f16)
    bn_stats_k<<<64, 256, 0, stream>>>(mtmp, stats, TT);
    bn_apply_k<<<blocks((long)BATCH * 64 * TT), EW, 0, stream>>>(mtmp, stats, bn_g, bn_b, h, m_h, TT);

    // 12) up1 64->1600 k15, pixel-shuffle x5 + leaky -> (b,320,5T) f16
    conv1d_wmma<64, 1600, 15, 7, 16, EPI_SHUF_LEAKY_F16, 5>
        <<<dim3(TT / 64, 100, BATCH), 128, 0, stream>>>(m_h, w16_up1, up1_b, nullptr, u5h, nullptr, TT);

    // 13) up2 320->256 k15 over 5T, pixel-shuffle x2 + leaky -> (b,128,10T) f16
    conv1d_wmma<320, 256, 15, 7, 16, EPI_SHUF_LEAKY_F16, 2>
        <<<dim3(TLEN5 / 64, 16, BATCH), 128, 0, stream>>>(u5h, w16_up2, up2_b, nullptr, u10h, nullptr, TLEN5);

    // 14) out conv 128->12 k15 over 10T -> d_out f32
    conv1d_wmma<128, 12, 15, 7, 16, EPI_F32, 1>
        <<<dim3(TLEN10 / 64, 1, BATCH), 128, 0, stream>>>(u10h, w16_out, out_b, (float*)d_out, nullptr, nullptr, TLEN10);
}
